// SparseGroupingLayer_35107062678370
// MI455X (gfx1250) — matrix operand order repaired and checked by reading, not compiled, not run on hardware
//
#include <hip/hip_runtime.h>
#include <hip/hip_bf16.h>

typedef _Float16 f16;
typedef f16 v16h __attribute__((ext_vector_type(16)));
typedef f16 v8h  __attribute__((ext_vector_type(8)));
typedef float v8f __attribute__((ext_vector_type(8)));

namespace {
constexpr int kB    = 8;
constexpr int kNin  = 3136;   // 56*56
constexpr int kNout = 784;    // 28*28
constexpr int kDin  = 96;
constexpr int kDout = 192;
constexpr int kHid  = 384;
constexpr float kEps = 1e-5f;
}

// ---------------------------------------------------------------------------
// Per-lane operand builders for v_wmma_f32_16x16x32_f16 (wave32).
// A (16x32 f16): lane<16 row=lane, elems 0..7 -> K 0..7, 8..15 -> K 16..23;
//                lanes 16..31 same rows, K shifted +8.
// B (32x16 f16): lane holds column N=lane&15; K mapping same as A.
// C/D (16x16 f32): elem r -> M = r + 8*(lane>=16), N = lane&15.
// ---------------------------------------------------------------------------
__device__ __forceinline__ v16h load_a_frag(const f16* __restrict__ ar, int kb, int hi) {
  v8h alo = *(const v8h*)(ar + kb + hi * 8);
  v8h ahi = *(const v8h*)(ar + kb + 16 + hi * 8);
  v16h a;
#pragma unroll
  for (int i = 0; i < 8; ++i) { a[i] = alo[i]; a[i + 8] = ahi[i]; }
  return a;
}

__device__ __forceinline__ v16h load_b_frag(const f16* __restrict__ wr, int kb, int hi) {
  v8h blo = *(const v8h*)(wr + kb + hi * 8);
  v8h bhi = *(const v8h*)(wr + kb + 16 + hi * 8);
  v16h b;
#pragma unroll
  for (int i = 0; i < 8; ++i) { b[i] = blo[i]; b[i + 8] = bhi[i]; }
  return b;
}

__device__ __forceinline__ float gelu_exact(float x) {
  return 0.5f * x * (1.0f + erff(x * 0.70710678118654752f));
}

// ---------------------------------------------------------------------------
// Register-blocked GEMM: Out[b] = A[b] * W^T (+bias) (gelu).
// One wave per block computes a 16 x (16*NT) output strip; A fragment is
// loaded once per k-step and reused across NT WMMAs.
// grid = (M/16, N/(16*NT), B), block = 32.
// FLAGS: bit0 = f16 output, bit1 = gelu.
// ---------------------------------------------------------------------------
template <int FLAGS, int NT>
__global__ void sgl_gemm(const f16* __restrict__ A, long long aStride, int lda,
                         const f16* __restrict__ W, long long wStride, int ldw,
                         void* __restrict__ Out, long long oStride, int ldo,
                         const float* __restrict__ bias, int K) {
  const int b = blockIdx.z;
  const int lane = threadIdx.x & 31;
  const int r    = lane & 15;
  const int hi   = lane >> 4;
  const f16* ar = A + (size_t)b * aStride + ((size_t)blockIdx.x * 16 + r) * lda;
  const f16* wr = W + (size_t)b * wStride + ((size_t)blockIdx.y * (16 * NT) + r) * ldw;

  v8f acc[NT];
#pragma unroll
  for (int t = 0; t < NT; ++t) acc[t] = v8f{0.f, 0.f, 0.f, 0.f, 0.f, 0.f, 0.f, 0.f};

#pragma unroll 2
  for (int kb = 0; kb < K; kb += 32) {
    const v16h a = load_a_frag(ar, kb, hi);
#pragma unroll
    for (int t = 0; t < NT; ++t) {
      const v16h bm = load_b_frag(wr + (size_t)t * 16 * ldw, kb, hi);
      acc[t] = __builtin_amdgcn_wmma_f32_16x16x32_f16(false, a, false, bm,
                                                      (short)0, acc[t], false, false);
    }
  }

  const int col = lane & 15;
  const size_t base = (size_t)b * oStride;
#pragma unroll
  for (int t = 0; t < NT; ++t) {
    const int n = blockIdx.y * (16 * NT) + t * 16 + col;
    const float bv = bias ? bias[n] : 0.f;
#pragma unroll
    for (int rr = 0; rr < 8; ++rr) {
      const int m = blockIdx.x * 16 + rr + hi * 8;
      float v = acc[t][rr] + bv;
      if (FLAGS & 2) v = gelu_exact(v);
      if (FLAGS & 1) ((f16*)Out)[base + (size_t)m * ldo + n]  = (f16)v;
      else           ((float*)Out)[base + (size_t)m * ldo + n] = v;
    }
  }
}

// ---------------------------------------------------------------------------
// Fused logits (WMMA) + mask/bias + row softmax.
// grid = (Nin/16, 1, B), block = 256 (8 waves).  16 rows x 784 cols in LDS.
// ---------------------------------------------------------------------------
__global__ void sgl_logits_softmax(const f16* __restrict__ Kf, const f16* __restrict__ Qf,
                                   const float* __restrict__ addm, const float* __restrict__ tau,
                                   float* __restrict__ aups) {
  __shared__ float tile[16 * 785];
  __shared__ float red[256];
  const int b  = blockIdx.z;
  const int mt = blockIdx.x;
  const float scale = __expf(tau[0]);
  const int wave = threadIdx.x >> 5;
  const int lane = threadIdx.x & 31;
  const int colL = lane & 15;
  const int hiL  = lane >> 4;
  const f16* ar = Kf + (size_t)b * kNin * kDin + ((size_t)mt * 16 + (lane & 15)) * kDin;
  const f16* Qb = Qf + (size_t)b * kNout * kDin;

  // A fragments for K=96 held in registers across all column tiles.
  v16h a0 = load_a_frag(ar, 0, hiL);
  v16h a1 = load_a_frag(ar, 32, hiL);
  v16h a2 = load_a_frag(ar, 64, hiL);

  for (int ct = wave; ct < kNout / 16; ct += 8) {
    const f16* wr = Qb + ((size_t)ct * 16 + (lane & 15)) * kDin;
    v8f acc = {0.f, 0.f, 0.f, 0.f, 0.f, 0.f, 0.f, 0.f};
    v16h b0 = load_b_frag(wr, 0, hiL);
    acc = __builtin_amdgcn_wmma_f32_16x16x32_f16(false, a0, false, b0, (short)0, acc, false, false);
    v16h b1 = load_b_frag(wr, 32, hiL);
    acc = __builtin_amdgcn_wmma_f32_16x16x32_f16(false, a1, false, b1, (short)0, acc, false, false);
    v16h b2 = load_b_frag(wr, 64, hiL);
    acc = __builtin_amdgcn_wmma_f32_16x16x32_f16(false, a2, false, b2, (short)0, acc, false, false);
    const int n = ct * 16 + colL;
#pragma unroll
    for (int r = 0; r < 8; ++r) {
      const int mL = r + hiL * 8;
      const int mG = mt * 16 + mL;
      tile[mL * 785 + n] = acc[r] * scale + addm[(size_t)mG * kNout + n];
    }
  }
  __syncthreads();
  const int row = threadIdx.x >> 4;   // 0..15
  const int c0  = threadIdx.x & 15;
  float mx = -3.0e38f;
  for (int c = c0; c < kNout; c += 16) mx = fmaxf(mx, tile[row * 785 + c]);
  red[threadIdx.x] = mx;
  __syncthreads();
  if (c0 == 0) {
    float m = red[row * 16];
    for (int i = 1; i < 16; ++i) m = fmaxf(m, red[row * 16 + i]);
    red[row * 16] = m;
  }
  __syncthreads();
  mx = red[row * 16];
  __syncthreads();
  float s = 0.f;
  for (int c = c0; c < kNout; c += 16) {
    float e = __expf(tile[row * 785 + c] - mx);
    tile[row * 785 + c] = e;
    s += e;
  }
  red[threadIdx.x] = s;
  __syncthreads();
  if (c0 == 0) {
    float t = 0.f;
    for (int i = 0; i < 16; ++i) t += red[row * 16 + i];
    red[row * 16] = t;
  }
  __syncthreads();
  const float inv = 1.0f / red[row * 16];
  const size_t obase = ((size_t)b * kNin + (size_t)mt * 16 + row) * kNout;
  for (int c = c0; c < kNout; c += 16) aups[obase + c] = tile[row * 785 + c] * inv;
}

// Deterministic column sums over N_in.
__global__ void sgl_col_sums(const float* __restrict__ aups, float* __restrict__ colsum) {
  const int b = blockIdx.y;
  const int c = blockIdx.x * 256 + threadIdx.x;
  if (c >= kNout) return;
  const float* p = aups + (size_t)b * kNin * kNout + c;
  float s = 0.f;
  for (int n = 0; n < kNin; ++n) s += p[(size_t)n * kNout];
  colsum[b * kNout + c] = s;
}

// ---------------------------------------------------------------------------
// a_down = a_ups / (colsum + 1e-8) (f32, coalesced) and LDS-tiled transpose
// to f16 aT[b][m][n].  64x64 tiles, LDS pitch 65 (64 banks -> conflict-free).
// grid = (Nin/64, ceil(Nout/64), B), block = 256.
// ---------------------------------------------------------------------------
__global__ void sgl_adown_tr(const float* __restrict__ aups, const float* __restrict__ colsum,
                             float* __restrict__ adown, f16* __restrict__ aT) {
  __shared__ float lds[64 * 65];
  const int b  = blockIdx.z;
  const int n0 = blockIdx.x * 64;
  const int m0 = blockIdx.y * 64;
  const float* ab = aups + (size_t)b * kNin * kNout;
  float* db = adown + (size_t)b * kNin * kNout;
  const float* cs = colsum + b * kNout;
#pragma unroll
  for (int j = 0; j < 16; ++j) {
    const int lin = j * 256 + threadIdx.x;
    const int nl = lin >> 6, ml = lin & 63;
    const int m = m0 + ml;
    float v = 0.f;
    if (m < kNout) {
      v = ab[(size_t)(n0 + nl) * kNout + m] / (cs[m] + 1e-8f);
      db[(size_t)(n0 + nl) * kNout + m] = v;
    }
    lds[ml * 65 + nl] = v;
  }
  __syncthreads();
  f16* tb = aT + (size_t)b * kNout * kNin;
#pragma unroll
  for (int j = 0; j < 16; ++j) {
    const int lin = j * 256 + threadIdx.x;
    const int ml = lin >> 6, nl = lin & 63;
    const int m = m0 + ml;
    if (m < kNout) tb[(size_t)m * kNin + n0 + nl] = (f16)lds[ml * 65 + nl];
  }
}

// Tiled f16 transpose: out[b][c][r] = in[b][r][c].  grid=(ceil(R/64),ceil(C/64),B).
__global__ void sgl_tr_f16(const f16* __restrict__ in, f16* __restrict__ out, int R, int C) {
  __shared__ f16 lds[64 * 65];
  const int b  = blockIdx.z;
  const int r0 = blockIdx.x * 64;
  const int c0 = blockIdx.y * 64;
  const f16* ib = in + (size_t)b * R * C;
  f16* ob = out + (size_t)b * R * C;
#pragma unroll
  for (int j = 0; j < 16; ++j) {
    const int lin = j * 256 + threadIdx.x;
    const int rl = lin >> 6, cl = lin & 63;
    const int r = r0 + rl, c = c0 + cl;
    f16 v = (f16)0.f;
    if (r < R && c < C) v = ib[(size_t)r * C + c];
    lds[cl * 65 + rl] = v;
  }
  __syncthreads();
#pragma unroll
  for (int j = 0; j < 16; ++j) {
    const int lin = j * 256 + threadIdx.x;
    const int cl = lin >> 6, rl = lin & 63;
    const int r = r0 + rl, c = c0 + cl;
    if (r < R && c < C) ob[(size_t)c * R + r] = lds[cl * 65 + rl];
  }
}

// Row LayerNorm -> f16 output. One wave per row, 8 rows per block.
__global__ void sgl_ln_f16(const float* __restrict__ in, const float* __restrict__ g,
                           const float* __restrict__ bt, f16* __restrict__ out,
                           int D, size_t rows) {
  const size_t row = (size_t)blockIdx.x * 8 + (threadIdx.x >> 5);
  if (row >= rows) return;
  const int lane = threadIdx.x & 31;
  const float* p = in + row * D;
  float s = 0.f, s2 = 0.f;
  for (int i = lane; i < D; i += 32) { float v = p[i]; s += v; s2 += v * v; }
  for (int o = 16; o > 0; o >>= 1) { s += __shfl_xor(s, o, 32); s2 += __shfl_xor(s2, o, 32); }
  const float mu = s / D;
  const float rr = rsqrtf(s2 / D - mu * mu + kEps);
  f16* q = out + row * D;
  for (int i = lane; i < D; i += 32) q[i] = (f16)((p[i] - mu) * rr * g[i] + bt[i]);
}

// x_out += LN(u); also mirror updated x_out to f16.
__global__ void sgl_add_ln(float* __restrict__ xout, const float* __restrict__ u,
                           const float* __restrict__ g, const float* __restrict__ bt,
                           f16* __restrict__ xcopy, size_t rows) {
  const size_t row = (size_t)blockIdx.x * 8 + (threadIdx.x >> 5);
  if (row >= rows) return;
  const int lane = threadIdx.x & 31;
  const float* p = u + row * kDout;
  float s = 0.f, s2 = 0.f;
  for (int i = lane; i < kDout; i += 32) { float v = p[i]; s += v; s2 += v * v; }
  for (int o = 16; o > 0; o >>= 1) { s += __shfl_xor(s, o, 32); s2 += __shfl_xor(s2, o, 32); }
  const float mu = s / kDout;
  const float rr = rsqrtf(s2 / kDout - mu * mu + kEps);
  float* xo = xout + row * kDout;
  f16* xc = xcopy + row * kDout;
  for (int i = lane; i < kDout; i += 32) {
    const float nv = (p[i] - mu) * rr * g[i] + bt[i];
    const float nx = xo[i] + nv;
    xo[i] = nx;
    xc[i] = (f16)nx;
  }
}

// Strided 3x3 conv (stride 2, pad 1) + LayerNorm -> x_out.  One block per (b, pixel).
__global__ void sgl_seed(const float* __restrict__ x, const float* __restrict__ cw,
                         const float* __restrict__ g, const float* __restrict__ bt,
                         float* __restrict__ xout) {
  __shared__ float patch[kDin * 9];
  __shared__ float vals[kDout];
  __shared__ float rs[64], rq[64];
  const int bp = blockIdx.x;
  const int b = bp / kNout;
  const int p = bp % kNout;
  const int oh = p / 28, ow = p % 28;
  const int tid = threadIdx.x;
  for (int t = tid; t < kDin * 9; t += kDout) {
    const int ic = t / 9, kp = t % 9;
    const int kh = kp / 3, kw = kp % 3;
    const int ih = oh * 2 - 1 + kh, iw = ow * 2 - 1 + kw;
    float v = 0.f;
    if (ih >= 0 && ih < 56 && iw >= 0 && iw < 56)
      v = x[((size_t)b * kNin + ih * 56 + iw) * kDin + ic];
    patch[t] = v;
  }
  __syncthreads();
  float acc = 0.f;
  const float* wp = cw + (size_t)tid * (kDin * 9);
  for (int t = 0; t < kDin * 9; ++t) acc += patch[t] * wp[t];
  vals[tid] = acc;
  __syncthreads();
  if (tid < 64) {
    const float v0 = vals[tid], v1 = vals[tid + 64], v2 = vals[tid + 128];
    rs[tid] = v0 + v1 + v2;
    rq[tid] = v0 * v0 + v1 * v1 + v2 * v2;
  }
  __syncthreads();
  for (int st = 32; st > 0; st >>= 1) {
    if (tid < st) { rs[tid] += rs[tid + st]; rq[tid] += rq[tid + st]; }
    __syncthreads();
  }
  const float mu = rs[0] / kDout;
  const float rr = rsqrtf(rq[0] / kDout - mu * mu + kEps);
  xout[(size_t)bp * kDout + tid] = (acc - mu) * rr * g[tid] + bt[tid];
}

__global__ void sgl_f32_to_f16(const float* __restrict__ in, f16* __restrict__ out, size_t n) {
  const size_t i = (size_t)blockIdx.x * blockDim.x + threadIdx.x;
  if (i < n) out[i] = (f16)in[i];
}

__global__ void sgl_build_add(const float* __restrict__ Mloc, const float* __restrict__ rpb,
                              const int* __restrict__ qidx, float* __restrict__ addm) {
  const int i = blockIdx.x * blockDim.x + threadIdx.x;
  if (i >= kNin * kNout) return;
  int q = qidx[i];
  if (q < 0) q += 9;   // torch-style wrap: -1 -> 8
  addm[i] = Mloc[i] + rpb[q];
}

// ---------------------------------------------------------------------------
extern "C" void kernel_launch(void* const* d_in, const int* in_sizes, int n_in,
                              void* d_out, int out_size, void* d_ws, size_t ws_size,
                              hipStream_t stream) {
  (void)in_sizes; (void)out_size;
  if (n_in < 21) return;
  const float* x        = (const float*)d_in[0];
  const float* conv_w   = (const float*)d_in[1];
  const float* q_w      = (const float*)d_in[2];
  const float* k_w      = (const float*)d_in[3];
  const float* v_w      = (const float*)d_in[4];
  const float* w1       = (const float*)d_in[5];
  const float* b1       = (const float*)d_in[6];
  const float* w2       = (const float*)d_in[7];
  const float* b2       = (const float*)d_in[8];
  const float* ln_in_g  = (const float*)d_in[9];
  const float* ln_in_b  = (const float*)d_in[10];
  const float* ln_out_g = (const float*)d_in[11];
  const float* ln_out_b = (const float*)d_in[12];
  const float* ln_at_g  = (const float*)d_in[13];
  const float* ln_at_b  = (const float*)d_in[14];
  const float* ln_ml_g  = (const float*)d_in[15];
  const float* ln_ml_b  = (const float*)d_in[16];
  const float* tau      = (const float*)d_in[17];
  const float* rpb      = (const float*)d_in[18];
  const float* Mloc     = (const float*)d_in[19];
  const int*   qidx     = (const int*)d_in[20];

  float* xout  = (float*)d_out;                          // [B, 784, 192]
  float* aups  = xout + (size_t)kB * kNout * kDout;      // [B, 3136, 784]
  float* adown = aups + (size_t)kB * kNin * kNout;       // [B, 3136, 784]

  char* wp = (char*)d_ws;
  auto alloc = [&](size_t bytes) -> void* {
    void* p = (void*)wp;
    wp += (bytes + 255) & ~(size_t)255;
    return p;
  };
  float* addm   = (float*)alloc((size_t)kNin * kNout * 4);
  f16*   xn     = (f16*)alloc((size_t)kB * kNin * kDin * 2);
  f16*   kf     = (f16*)alloc((size_t)kB * kNin * kDin * 2);
  f16*   vtmp   = (f16*)alloc((size_t)kB * kNin * kDout * 2);
  f16*   vT     = (f16*)alloc((size_t)kB * kDout * kNin * 2);
  f16*   lnxo   = (f16*)alloc((size_t)kB * kNout * kDout * 2);
  f16*   qf     = (f16*)alloc((size_t)kB * kNout * kDin * 2);
  float* colsum = (float*)alloc((size_t)kB * kNout * 4);
  f16*   aT     = (f16*)alloc((size_t)kB * kNout * kNin * 2);
  float* upd    = (float*)alloc((size_t)kB * kNout * kDout * 4);
  f16*   xo16   = (f16*)alloc((size_t)kB * kNout * kDout * 2);
  f16*   hf     = (f16*)alloc((size_t)kB * kNout * kHid * 2);
  float* h2     = (float*)alloc((size_t)kB * kNout * kDout * 4);
  f16*   qw16   = (f16*)alloc((size_t)kDin * kDout * 2);
  f16*   kw16   = (f16*)alloc((size_t)kDin * kDin * 2);
  f16*   vw16   = (f16*)alloc((size_t)kDout * kDin * 2);
  f16*   w116   = (f16*)alloc((size_t)kHid * kDout * 2);
  f16*   w216   = (f16*)alloc((size_t)kDout * kHid * 2);
  if ((size_t)(wp - (char*)d_ws) > ws_size) return;

  auto conv16 = [&](const float* src, f16* dst, size_t n) {
    sgl_f32_to_f16<<<(unsigned)((n + 255) / 256), 256, 0, stream>>>(src, dst, n);
  };
  conv16(q_w, qw16, (size_t)kDin * kDout);
  conv16(k_w, kw16, (size_t)kDin * kDin);
  conv16(v_w, vw16, (size_t)kDout * kDin);
  conv16(w1, w116, (size_t)kHid * kDout);
  conv16(w2, w216, (size_t)kDout * kHid);

  sgl_build_add<<<(kNin * kNout + 255) / 256, 256, 0, stream>>>(Mloc, rpb, qidx, addm);
  sgl_seed<<<kB * kNout, kDout, 0, stream>>>(x, conv_w, ln_out_g, ln_out_b, xout);
  sgl_ln_f16<<<(unsigned)(((size_t)kB * kNin + 7) / 8), 256, 0, stream>>>(
      x, ln_in_g, ln_in_b, xn, kDin, (size_t)kB * kNin);

  // k = LN(x) @ k_w.T  -> f16 [B, 3136, 96]
  sgl_gemm<1, 2><<<dim3(kNin / 16, kDin / 32, kB), 32, 0, stream>>>(
      xn, (long long)kNin * kDin, kDin, kw16, 0, kDin,
      kf, (long long)kNin * kDin, kDin, nullptr, kDin);
  // v = LN(x) @ v_w.T -> f16 [B, 3136, 192], then tiled transpose -> vT [B, 192, 3136]
  sgl_gemm<1, 4><<<dim3(kNin / 16, kDout / 64, kB), 32, 0, stream>>>(
      xn, (long long)kNin * kDin, kDin, vw16, 0, kDin,
      vtmp, (long long)kNin * kDout, kDout, nullptr, kDin);
  sgl_tr_f16<<<dim3(kNin / 64, kDout / 64, kB), 256, 0, stream>>>(vtmp, vT, kNin, kDout);

  for (int it = 0; it < 3; ++it) {
    // q = LN(x_out) @ q_w.T -> f16 [B, 784, 96]
    sgl_ln_f16<<<(unsigned)(((size_t)kB * kNout + 7) / 8), 256, 0, stream>>>(
        xout, ln_out_g, ln_out_b, lnxo, kDout, (size_t)kB * kNout);
    sgl_gemm<1, 2><<<dim3(kNout / 16, kDin / 32, kB), 32, 0, stream>>>(
        lnxo, (long long)kNout * kDout, kDout, qw16, 0, kDout,
        qf, (long long)kNout * kDin, kDin, nullptr, kDout);
    // logits + softmax -> a_ups
    sgl_logits_softmax<<<dim3(kNin / 16, 1, kB), 256, 0, stream>>>(kf, qf, addm, tau, aups);
    // column renorm -> a_down + tiled f16 transpose aT
    sgl_col_sums<<<dim3((kNout + 255) / 256, kB), 256, 0, stream>>>(aups, colsum);
    sgl_adown_tr<<<dim3(kNin / 64, (kNout + 63) / 64, kB), 256, 0, stream>>>(
        aups, colsum, adown, aT);
    // updates = a_down^T @ v  -> f32 [B, 784, 192]
    sgl_gemm<0, 4><<<dim3(kNout / 16, kDout / 64, kB), 32, 0, stream>>>(
        aT, (long long)kNout * kNin, kNin, vT, (long long)kDout * kNin, kNin,
        upd, (long long)kNout * kDout, kDout, nullptr, kNin);
    sgl_add_ln<<<(unsigned)(((size_t)kB * kNout + 7) / 8), 256, 0, stream>>>(
        xout, upd, ln_at_g, ln_at_b, xo16, (size_t)kB * kNout);
    // MLP
    sgl_gemm<3, 4><<<dim3(kNout / 16, kHid / 64, kB), 32, 0, stream>>>(
        xo16, (long long)kNout * kDout, kDout, w116, 0, kDout,
        hf, (long long)kNout * kHid, kHid, b1, kDout);
    sgl_gemm<0, 4><<<dim3(kNout / 16, kDout / 64, kB), 32, 0, stream>>>(
        hf, (long long)kNout * kHid, kHid, w216, 0, kHid,
        h2, (long long)kNout * kDout, kDout, b2, kHid);
    sgl_add_ln<<<(unsigned)(((size_t)kB * kNout + 7) / 8), 256, 0, stream>>>(
        xout, h2, ln_ml_g, ln_ml_b, xo16, (size_t)kB * kNout);
  }
}
